// Hippocampus_11699490914624
// MI455X (gfx1250) — matrix operand order repaired and checked by reading, not compiled, not run on hardware
//
#include <hip/hip_runtime.h>
#include <stdint.h>

// ---------------- types / constants ----------------
typedef __attribute__((ext_vector_type(16))) _Float16 v16h;
typedef __attribute__((ext_vector_type(8)))  _Float16 v8h;
typedef __attribute__((ext_vector_type(8)))  float    v8f;
typedef int v4i __attribute__((vector_size(16)));

#define D_SEM     1024
#define CAPACITY  65536
#define BATCH     256
#define TOPK      16
#define SPARSE_K  512
#define EPSN      1e-8f
#define NEG_INF   (-3.402823466e38f)
#define NSLICES   256          // capacity slices, 256 rows each
#define REC_ELEMS (BATCH * TOPK * D_SEM)

// ---- gfx1250 async global->LDS path (probe-verified signature; falls back if absent) ----
#ifdef __has_builtin
#if __has_builtin(__builtin_amdgcn_global_load_async_to_lds_b128)
#define HAS_ASYNC_LDS 1
#endif
#endif
#ifndef HAS_ASYNC_LDS
#define HAS_ASYNC_LDS 0
#endif

#if HAS_ASYNC_LDS
// signature (from hipcc diagnostic): (AS1 v4i* gsrc, AS3 v4i* ldst, imm offset, imm cpol)
#define GL_ASYNC_B128(gp, lp)                                                        \
    __builtin_amdgcn_global_load_async_to_lds_b128(                                  \
        (__attribute__((address_space(1))) v4i*)(void*)(gp),                         \
        (__attribute__((address_space(3))) v4i*)(void*)(lp), 0, 0)
#if __has_builtin(__builtin_amdgcn_s_wait_asynccnt)
#define WAIT_ASYNC(n) __builtin_amdgcn_s_wait_asynccnt(n)
#else
#define WAIT_ASYNC(n) asm volatile("s_wait_asynccnt %0" ::"i"(n) : "memory")
#endif
#endif

// monotone unsigned key for float ordering (radix select)
static __device__ inline unsigned f2mono(float f) {
    unsigned u = __float_as_uint(f);
    return (u & 0x80000000u) ? ~u : (u | 0x80000000u);
}

// ---- WMMA f16 helpers (layouts per CDNA5 ISA 7.12.2) ----
// A fragment: 16x32 (MxK) from row-major f16 tile. lane m = lane&15.
// lanes 0-15: K = {0..7, 16..23}; lanes 16-31: K = {8..15, 24..31}.
static __device__ inline v16h load_a_frag(const _Float16* base, int lda, int lane) {
    int m = lane & 15, hi = lane >> 4;
    const _Float16* p = base + m * lda + 8 * hi;
    v8h c0 = *(const v8h*)p;
    v8h c1 = *(const v8h*)(p + 16);
    return __builtin_shufflevector(c0, c1, 0,1,2,3,4,5,6,7,8,9,10,11,12,13,14,15);
}
// B fragment: 32x16 (KxN). Source tile is row-major [n][k] (K contiguous).
static __device__ inline v16h load_b_frag(const _Float16* base, int ldb, int lane) {
    int n = lane & 15, hi = lane >> 4;
    const _Float16* p = base + n * ldb + 8 * hi;
    v8h c0 = *(const v8h*)p;
    v8h c1 = *(const v8h*)(p + 16);
    return __builtin_shufflevector(c0, c1, 0,1,2,3,4,5,6,7,8,9,10,11,12,13,14,15);
}
static __device__ inline v8f wmma_f16(v16h a, v16h b, v8f c) {
    return __builtin_amdgcn_wmma_f32_16x16x32_f16(false, a, false, b, (short)0, c, false, false);
}

// ================= Phase 0: DG projection + top-512 sparsify + normalize =================
// grid 16 x 256 threads; block b owns query rows [16b, 16b+16)
#define DG_LDS 164992
__global__ void hippo_dg(const float* __restrict__ query, const float* __restrict__ Wdg,
                         _Float16* __restrict__ keyn) {
    extern __shared__ char smem[];
    _Float16* Aall  = (_Float16*)smem;              // [16][1024] f16
    _Float16* Bc    = (_Float16*)(smem + 32768);    // [1024][32] f16
    float*    res   = (float*)(smem + 98304);       // [16][1024] f32
    int*      cnt   = (int*)(smem + 163840);        // [16]
    float*    psum  = (float*)(smem + 163904);      // [16][16]
    float*    scl   = (float*)(smem + 164928);      // [16]

    const int t = threadIdx.x;
    const int w = t >> 5, lane = t & 31;
    const int hi = lane >> 4, l16 = lane & 15;
    const int q0 = blockIdx.x * 16;

    for (int i = t; i < 16 * D_SEM; i += 256)
        Aall[i] = (_Float16)query[(size_t)(q0 + (i >> 10)) * D_SEM + (i & 1023)];
    __syncthreads();

    v8f acc[8]; v8f z = {};
#pragma unroll
    for (int i = 0; i < 8; ++i) acc[i] = z;

    for (int kc = 0; kc < 32; ++kc) {
        // stage B: W_dg rows (N) x 32 K-cols, f32 -> f16
#pragma unroll
        for (int rr = 0; rr < 4; ++rr) {
            int n = t * 4 + rr;
            const float4* s4 = (const float4*)(Wdg + (size_t)n * D_SEM + kc * 32);
            _Float16* dst = Bc + n * 32;
#pragma unroll
            for (int c = 0; c < 8; ++c) {
                float4 f = s4[c];
                dst[4*c+0] = (_Float16)f.x; dst[4*c+1] = (_Float16)f.y;
                dst[4*c+2] = (_Float16)f.z; dst[4*c+3] = (_Float16)f.w;
            }
        }
        __syncthreads();
        // per-WMMA B-fragment load: stays within register budget (no scratch spills)
        v16h a = load_a_frag(Aall + kc * 32, D_SEM, lane);
#pragma unroll
        for (int nt = 0; nt < 8; ++nt) {
            v16h b = load_b_frag(Bc + (128 * w + 16 * nt) * 32, 32, lane);
            acc[nt] = wmma_f16(a, b, acc[nt]);
        }
        __syncthreads();
    }
#pragma unroll
    for (int nt = 0; nt < 8; ++nt)
#pragma unroll
        for (int v = 0; v < 8; ++v)
            res[(v + 8 * hi) * D_SEM + 128 * w + 16 * nt + l16] = acc[nt][v];
    __syncthreads();

    // radix select 512th largest per row; 16 threads per row
    const int r = t >> 4, sub = t & 15;
    float vals[64];
#pragma unroll
    for (int i = 0; i < 64; ++i) vals[i] = res[r * D_SEM + sub + 16 * i];

    unsigned T = 0;
    for (int bit = 31; bit >= 0; --bit) {
        if (sub == 0) cnt[r] = 0;
        __syncthreads();
        unsigned cand = T | (1u << bit);
        int c = 0;
#pragma unroll
        for (int i = 0; i < 64; ++i) c += (f2mono(vals[i]) >= cand) ? 1 : 0;
        atomicAdd(&cnt[r], c);
        __syncthreads();
        if (cnt[r] >= SPARSE_K) T = cand;
        __syncthreads();
    }

    float s = 0.f;
#pragma unroll
    for (int i = 0; i < 64; ++i) { float f = vals[i]; if (f2mono(f) >= T) s += f * f; }
    psum[r * 16 + sub] = s;
    __syncthreads();
    if (sub == 0) {
        float tot = 0.f;
        for (int i = 0; i < 16; ++i) tot += psum[r * 16 + i];   // fixed order -> deterministic
        scl[r] = 1.0f / fmaxf(sqrtf(tot), EPSN);
    }
    __syncthreads();
    float sc = scl[r];
#pragma unroll
    for (int i = 0; i < 64; ++i) {
        int col = sub + 16 * i;
        float f = vals[i];
        keyn[(size_t)(q0 + r) * D_SEM + col] = (_Float16)((f2mono(f) >= T) ? f * sc : 0.0f);
    }
}

// ================= Phase 1b: W_ca1 f32 -> f16 =================
__global__ void hippo_wconv(const float* __restrict__ W, _Float16* __restrict__ Wh) {
    size_t i = ((size_t)blockIdx.x * 256 + threadIdx.x) * 4;
    float4 f = *(const float4*)(W + i);
    Wh[i + 0] = (_Float16)f.x; Wh[i + 1] = (_Float16)f.y;
    Wh[i + 2] = (_Float16)f.z; Wh[i + 3] = (_Float16)f.w;
}

// ================= Phase 2: cosine-sim GEMM + slice-local top-16 =================
// grid 256 x 256 threads; block = all 256 queries x 256-row key slice (keys read ONCE total)
// A (key_n f16) staged via async global->LDS double buffer; keys converted+norm'd in registers.
// dyn LDS: Ac[2] 32768 | Bc 4096 | simb(T) 65536 | topv 16384 | topi 16384 | pS 1024 | rnorm 256
#define SIM_LDS 136448
__global__ void hippo_sim(const _Float16* __restrict__ keyn, const float* __restrict__ keys,
                          float* __restrict__ candv, int* __restrict__ candi) {
    extern __shared__ char smem[];
    _Float16* Ac    = (_Float16*)smem;               // [2][256][32]
    _Float16* Bc    = (_Float16*)(smem + 32768);     // [64][32]
    float*    simb  = (float*)(smem + 36864);        // [64][256]  (transposed: [n][m])
    float*    topv  = (float*)(smem + 102400);       // [256][16]
    int*      topi  = (int*)(smem + 118784);         // [256][16]
    float*    pS    = (float*)(smem + 135168);       // [64][4]
    float*    rnorm = (float*)(smem + 136192);       // [64]

    const int t = threadIdx.x;
    const int w = t >> 5, lane = t & 31;
    const int hi = lane >> 4, l16 = lane & 15;
    const int slice = blockIdx.x;
    const int br = t >> 2, c0 = (t & 3) * 8;   // staging: 4 threads per row, 16B pieces

#pragma unroll
    for (int j = 0; j < 16; ++j) { topv[t * 16 + j] = NEG_INF; topi[t * 16 + j] = 0; }

    const _Float16* arow = keyn + (size_t)t * D_SEM;

    for (int ntile = 0; ntile < 4; ++ntile) {
        const int nbase = slice * 256 + ntile * 64;
        const float* brow = keys + (size_t)(nbase + br) * D_SEM + c0;

        v8f acc[2][4]; v8f z = {};
#pragma unroll
        for (int mt = 0; mt < 2; ++mt)
#pragma unroll
            for (int nt = 0; nt < 4; ++nt) acc[mt][nt] = z;
        float ssq = 0.f;

        // -------- prologue: chunk 0 --------
        float4 bbuf0, bbuf1;
        {
            const float4* s4 = (const float4*)brow;
            bbuf0 = s4[0]; bbuf1 = s4[1];
        }
#if HAS_ASYNC_LDS
        // A pieces: 1024 x 16B per chunk; thread handles 4 pieces (row = p>>2, 16B col = p&3)
#pragma unroll
        for (int i = 0; i < 4; ++i) {
            int p = t + 256 * i, ar = p >> 2, ach = (p & 3) * 8;
            GL_ASYNC_B128(keyn + (size_t)ar * D_SEM + ach, Ac + ar * 32 + ach);
        }
#else
        v16h abuf0, abuf1;
        { const v16h* s = (const v16h*)arow; abuf0 = s[0]; abuf1 = s[1]; }
#endif

        for (int kc = 0; kc < 32; ++kc) {
            _Float16* AcB = Ac + (kc & 1) * 8192;
            // -------- consume B register buffer: norms + convert + store --------
            {
                float4 f0 = bbuf0, f1 = bbuf1;
                ssq += f0.x*f0.x + f0.y*f0.y + f0.z*f0.z + f0.w*f0.w
                     + f1.x*f1.x + f1.y*f1.y + f1.z*f1.z + f1.w*f1.w;
                v8h h;
                h[0]=(_Float16)f0.x; h[1]=(_Float16)f0.y; h[2]=(_Float16)f0.z; h[3]=(_Float16)f0.w;
                h[4]=(_Float16)f1.x; h[5]=(_Float16)f1.y; h[6]=(_Float16)f1.z; h[7]=(_Float16)f1.w;
                *(v8h*)(Bc + br * 32 + c0) = h;
            }
#if !HAS_ASYNC_LDS
            { v16h* d = (v16h*)(AcB + t * 32); d[0] = abuf0; d[1] = abuf1; }
#endif
            // -------- issue next chunk (hidden under this chunk's WMMAs) --------
#if HAS_ASYNC_LDS
            if (kc < 31) {
                _Float16* AcN = Ac + ((kc + 1) & 1) * 8192;
#pragma unroll
                for (int i = 0; i < 4; ++i) {
                    int p = t + 256 * i, ar = p >> 2, ach = (p & 3) * 8;
                    GL_ASYNC_B128(keyn + (size_t)ar * D_SEM + (kc + 1) * 32 + ach,
                                  AcN + ar * 32 + ach);
                }
                WAIT_ASYNC(4);     // in-order: chunk kc's 4 loads have landed
            } else {
                WAIT_ASYNC(0);
            }
#else
            if (kc < 31) {
                const v16h* s = (const v16h*)(arow + (kc + 1) * 32);
                abuf0 = s[0]; abuf1 = s[1];
            }
#endif
            if (kc < 31) {
                const float* bsrc = brow + (kc + 1) * 32;
                const float4* s4 = (const float4*)bsrc;
                bbuf0 = s4[0]; bbuf1 = s4[1];
                if (kc < 30) __builtin_prefetch(bsrc + 32, 0, 1);   // global_prefetch_b8
            }
            __syncthreads();
            // -------- fragments once, then the 8-WMMA chain --------
            v16h a0 = load_a_frag(AcB + (32 * w) * 32, 32, lane);
            v16h a1 = load_a_frag(AcB + (32 * w + 16) * 32, 32, lane);
#pragma unroll
            for (int nt = 0; nt < 4; ++nt) {
                v16h b = load_b_frag(Bc + (16 * nt) * 32, 32, lane);
                acc[0][nt] = wmma_f16(a0, b, acc[0][nt]);
                acc[1][nt] = wmma_f16(a1, b, acc[1][nt]);
            }
            __syncthreads();
        }
        // finalize bank-row norms (deterministic 4-way sum)
        pS[br * 4 + (t & 3)] = ssq;
        __syncthreads();
        if (t < 64) {
            float tot = pS[t*4+0] + pS[t*4+1] + pS[t*4+2] + pS[t*4+3];
            rnorm[t] = 1.0f / fmaxf(sqrtf(tot), EPSN);
        }
        // dump sim tile transposed [n][m]: contiguous per-lane -> b128 stores
#pragma unroll
        for (int mt = 0; mt < 2; ++mt)
#pragma unroll
            for (int nt = 0; nt < 4; ++nt) {
                float* dst = simb + (16 * nt + l16) * 256 + 32 * w + 16 * mt + 8 * hi;
#pragma unroll
                for (int v = 0; v < 8; ++v) dst[v] = acc[mt][nt][v];
            }
        __syncthreads();
        // per-query top-16 insertion (thread t owns query t)
        float* tv = topv + t * 16; int* ti = topi + t * 16;
        for (int j = 0; j < 64; ++j) {
            float sv = simb[j * 256 + t] * rnorm[j];
            if (sv > tv[15]) {
                int pos = 15;
                while (pos > 0 && tv[pos - 1] < sv) {
                    tv[pos] = tv[pos - 1]; ti[pos] = ti[pos - 1]; --pos;
                }
                tv[pos] = sv; ti[pos] = nbase + j;
            }
        }
        __syncthreads();
    }
#pragma unroll
    for (int j = 0; j < 16; ++j) {
        candv[(size_t)slice * 4096 + t * 16 + j] = topv[t * 16 + j];
        candi[(size_t)slice * 4096 + t * 16 + j] = topi[t * 16 + j];
    }
}

// ================= Phase 3: global top-16 merge + novelty + value gather =================
__global__ void hippo_merge(const float* __restrict__ candv, const int* __restrict__ candi,
                            const float* __restrict__ values, _Float16* __restrict__ A2,
                            float* __restrict__ out) {
    __shared__ float cv[4096];
    __shared__ int   ci[4096];
    __shared__ float redv[256];
    __shared__ int   redi[256];
    __shared__ float outv[16];
    __shared__ int   outi[16];
    const int t = threadIdx.x, q = blockIdx.x;

    for (int i = t; i < 4096; i += 256) {
        int s = i >> 4, j = i & 15;
        cv[i] = candv[(size_t)s * 4096 + q * 16 + j];
        ci[i] = candi[(size_t)s * 4096 + q * 16 + j];
    }
    __syncthreads();

    for (int r = 0; r < 16; ++r) {
        float bv = NEG_INF; int bi = 0;
#pragma unroll
        for (int k = 0; k < 16; ++k) {
            int i = t + 256 * k;
            if (cv[i] > bv) { bv = cv[i]; bi = i; }
        }
        redv[t] = bv; redi[t] = bi;
        __syncthreads();
        for (int st = 128; st > 0; st >>= 1) {
            if (t < st && redv[t + st] > redv[t]) { redv[t] = redv[t + st]; redi[t] = redi[t + st]; }
            __syncthreads();
        }
        if (t == 0) { int sl = redi[0]; outv[r] = cv[sl]; outi[r] = ci[sl]; cv[sl] = NEG_INF; }
        __syncthreads();
    }
    if (t == 0) {
        float mv = outv[0];
        float nv = 1.0f - fminf(fmaxf(mv, -1.0f), 1.0f);
        out[REC_ELEMS + q] = fminf(fmaxf(nv, 0.0f), 1.0f);
    }
    // gather values rows -> f16 A matrix for CA1 GEMM
    for (int j = 0; j < 16; ++j) {
        int row = outi[j];
        for (int c = t; c < D_SEM; c += 256)
            A2[(size_t)(q * 16 + j) * D_SEM + c] = (_Float16)values[(size_t)row * D_SEM + c];
    }
}

// ================= Phase 4: recalled = gathered_values @ W_ca1^T (4096x1024x1024) =================
// grid 1024 (64 M-blocks x 16 N-blocks) x 256 threads; async double-buffered f16 staging
__global__ void hippo_ca1(const _Float16* __restrict__ A2, const _Float16* __restrict__ Wh,
                          float* __restrict__ out) {
    __shared__ _Float16 Ac[2][64 * 32];
    __shared__ _Float16 Bc[2][64 * 32];
    const int t = threadIdx.x, w = t >> 5, lane = t & 31;
    const int hi = lane >> 4, l16 = lane & 15;
    const int mblk = blockIdx.x >> 4, nblk = blockIdx.x & 15;
    const int mrow0 = (w & 3) * 16, ncol0 = (w >> 2) * 32;
    const int r = t >> 2, c0 = (t & 3) * 8;

    const _Float16* asrc = A2 + (size_t)(mblk * 64 + r) * D_SEM + c0;
    const _Float16* bsrc = Wh + (size_t)(nblk * 64 + r) * D_SEM + c0;

    v8f acc[2]; v8f z = {};
    acc[0] = z; acc[1] = z;

#if HAS_ASYNC_LDS
    GL_ASYNC_B128(asrc, &Ac[0][r * 32 + c0]);
    GL_ASYNC_B128(bsrc, &Bc[0][r * 32 + c0]);
#else
    v8h abuf = *(const v8h*)asrc;
    v8h bbuf = *(const v8h*)bsrc;
#endif

    for (int kc = 0; kc < 32; ++kc) {
        const int cur = kc & 1;
#if HAS_ASYNC_LDS
        if (kc < 31) {
            GL_ASYNC_B128(asrc + (kc + 1) * 32, &Ac[cur ^ 1][r * 32 + c0]);
            GL_ASYNC_B128(bsrc + (kc + 1) * 32, &Bc[cur ^ 1][r * 32 + c0]);
            WAIT_ASYNC(2);     // in-order: chunk kc's 2 loads have landed
        } else {
            WAIT_ASYNC(0);
        }
#else
        *(v8h*)(&Ac[cur][r * 32 + c0]) = abuf;
        *(v8h*)(&Bc[cur][r * 32 + c0]) = bbuf;
        if (kc < 31) {
            abuf = *(const v8h*)(asrc + (kc + 1) * 32);
            bbuf = *(const v8h*)(bsrc + (kc + 1) * 32);
        }
#endif
        __syncthreads();
        v16h a = load_a_frag(&Ac[cur][mrow0 * 32], 32, lane);
#pragma unroll
        for (int h = 0; h < 2; ++h) {
            v16h b = load_b_frag(&Bc[cur][(ncol0 + 16 * h) * 32], 32, lane);
            acc[h] = wmma_f16(a, b, acc[h]);
        }
        __syncthreads();
    }
#pragma unroll
    for (int h = 0; h < 2; ++h)
#pragma unroll
        for (int v = 0; v < 8; ++v) {
            int m = mblk * 64 + mrow0 + v + 8 * hi;
            int n = nblk * 64 + ncol0 + 16 * h + l16;
            out[(size_t)m * D_SEM + n] = acc[h][v];
        }
}

// ================= host launcher =================
extern "C" void kernel_launch(void* const* d_in, const int* in_sizes, int n_in,
                              void* d_out, int out_size, void* d_ws, size_t ws_size,
                              hipStream_t stream) {
    const float* query  = (const float*)d_in[0];
    const float* Wdg    = (const float*)d_in[1];
    const float* Wca1   = (const float*)d_in[2];
    const float* keys   = (const float*)d_in[3];
    const float* values = (const float*)d_in[4];
    float* out = (float*)d_out;

    char* ws = (char*)d_ws;
    _Float16* keyn  = (_Float16*)(ws);                                   // 512 KB
    float*    candv = (float*)(ws + (512u << 10));                       // 4 MB
    int*      candi = (int*)(ws + (512u << 10) + (4u << 20));            // 4 MB
    _Float16* A2    = (_Float16*)(ws + (512u << 10) + (8u << 20));       // 8 MB
    _Float16* Wh    = (_Float16*)(ws + (512u << 10) + (16u << 20));      // 2 MB

    hippo_dg   <<<16,   256, DG_LDS,  stream>>>(query, Wdg, keyn);
    hippo_wconv<<<1024, 256, 0,       stream>>>(Wca1, Wh);
    hippo_sim  <<<NSLICES, 256, SIM_LDS, stream>>>(keyn, keys, candv, candi);
    hippo_merge<<<BATCH, 256, 0,      stream>>>(candv, candi, values, A2, out);
    hippo_ca1  <<<1024, 256, 0,       stream>>>(A2, Wh, out);
}
